// MambawDecoder_78950088835617
// MI455X (gfx1250) — matrix operand order repaired and checked by reading, not compiled, not run on hardware
//
#include <hip/hip_runtime.h>
#include <math.h>

// ---------------- model dimensions (from reference) ----------------
#define BATCH     4
#define SEQLEN    1024
#define DINPUT    64
#define DMODEL    256
#define DOUTPUT   1000
#define DSTATE    64
#define DCONV     4
#define NLAYERS   2
#define DINNER    512                    // EXPAND * DMODEL
#define HEADDIM   64
#define NHEADS    8                      // DINNER / HEADDIM
#define CONVDIM   640                    // DINNER + 2*DSTATE
#define DINPROJ   1160                   // 2*DINNER + 2*DSTATE + NHEADS
#define L1DIM     350
#define L2DIM     400

typedef __attribute__((ext_vector_type(16))) _Float16 v16h;
typedef __attribute__((ext_vector_type(8)))  float    v8f;

__device__ __forceinline__ float sigmoidf_(float x) { return 1.0f / (1.0f + __expf(-x)); }
__device__ __forceinline__ float siluf_(float x)    { return x * sigmoidf_(x); }

// ---------------------------------------------------------------------------
// WMMA GEMM:  C[M,N] = act( A[M,K] (row-major, stride lda) * W[K,N] + bias )
//
// Block = 128 threads (4 waves). Block tile = 32(M) x 64(N): two 16-row A
// subtiles share one staged B tile, so each wave issues TWO WMMAs per K-step
// against the same B fragment (halves B staging per FLOP). K-loop steps by 32
// with DOUBLE-BUFFERED LDS: iteration i reads fragments of tile i while
// issuing the global loads + LDS stores for tile i+1; one barrier per step.
//
// LDS tiles are f16, PRE-SWIZZLED into the CDNA5 WMMA fragment layout
// (ISA 7.12.2) so each lane reads a 16-half fragment as one contiguous 32-byte
// LDS read (2x ds_load_b128):
//   A 16x32 : element (m,k) -> tile[m + 16*((k>>3)&1)][(k&7) + 8*((k>>4)&1)]
//   B 32x64 : element (k,n) -> sB[n>>4][(n&15) + 16*((k>>3)&1)][(k&7) + 8*((k>>4)&1)]
//
// Bounds: scalar-uniform "full tile" tests select an unguarded fast path
// (immediate-offset loads); the guarded path (clamped index + cndmask,
// branch-free) runs only on edge tiles. No early returns -> all waves hit
// every barrier and EXEC is all-ones at the WMMA.
// ---------------------------------------------------------------------------
__global__ void wmma_gemm_kernel(const float* __restrict__ A, int lda,
                                 const float* __restrict__ W,      // K x N
                                 const float* __restrict__ bias,   // N or null
                                 float* __restrict__ C, int ldc,
                                 int M, int N, int K, int act)
{
    const int t        = threadIdx.x;       // 0..127
    const int wave     = t >> 5;            // 0..3
    const int lane     = t & 31;
    const int half_sel = lane >> 4;
    const int l16      = lane & 15;

    const int m0   = blockIdx.x * 32;       // two 16-row subtiles
    const int nblk = blockIdx.y * 64;

    __shared__ alignas(32) _Float16 sA[2][2][32][16];     // [buf][msub] 2x2 KB
    __shared__ alignas(32) _Float16 sB[2][4][32][16];     // [buf][nsub] 2x4 KB

    // --- staging coordinates (fixed per thread) ---
    // A: thread -> within-tile row a_m = t>>3 (0..15), k-quad = (t&7)*4.
    //    Each thread stages that row in BOTH 16-row subtiles (8 elems total).
    const int a_m    = t >> 3;
    const int a_kq   = (t & 7) * 4;
    const int a_mg0  = m0 + a_m;
    const int a_mg1  = m0 + 16 + a_m;
    const bool a_ok0 = (a_mg0 < M);
    const bool a_ok1 = (a_mg1 < M);
    const size_t a_row0 = (size_t)(a_ok0 ? a_mg0 : 0) * lda;
    const size_t a_row1 = (size_t)(a_ok1 ? a_mg1 : 0) * lda;
    // A LDS slot (within a 512-half tile): quad of consecutive k -> 4
    // consecutive halfs (b64-packable)
    const int a_slot = (a_m + 16 * ((a_kq >> 3) & 1)) * 16 +
                       (a_kq & 7) + 8 * ((a_kq >> 4) & 1);
    const size_t a_tile_stride = 32 * 16;         // halves per A subtile
    const size_t a_buf_stride  = 2 * 32 * 16;     // halves per A buffer

    // B: thread -> k-row = t>>2 (0..31), n-chunk = (t&3)*16  (2048 elems)
    const int b_k  = t >> 2;
    const int b_nc = t & 3;
    const int b_slot = (b_nc * 32 + 16 * ((b_k >> 3) & 1)) * 16 + (b_k & 7) + 8 * ((b_k >> 4) & 1);
    const size_t b_buf_stride = 4 * 32 * 16;      // halves per B buffer

    const bool fullM = (m0 + 32 <= M);            // scalar-uniform
    const bool fullN = (nblk + 64 <= N);          // scalar-uniform

    _Float16* const sAflat = &sA[0][0][0][0];
    _Float16* const sBflat = &sB[0][0][0][0];

    // ---- tile stager (fast unguarded path vs guarded edge path) ----
    auto stage = [&](int buf, int k0) {
        const bool fullK = (k0 + 32 <= K);
        _Float16* aw0 = sAflat + (size_t)buf * a_buf_stride + a_slot;
        _Float16* aw1 = aw0 + a_tile_stride;
        _Float16* bw  = sBflat + (size_t)buf * b_buf_stride + b_slot;
        if (fullM && fullN && fullK) {
            // fast path: base pointers + immediate offsets, no guards
            const float* ap0 = A + a_row0 + k0 + a_kq;
            const float* ap1 = A + a_row1 + k0 + a_kq;
#pragma unroll
            for (int j = 0; j < 4; ++j) aw0[j] = (_Float16)ap0[j];
#pragma unroll
            for (int j = 0; j < 4; ++j) aw1[j] = (_Float16)ap1[j];
            const float* wp = W + (size_t)(k0 + b_k) * N + nblk + b_nc * 16;
#pragma unroll
            for (int j = 0; j < 16; ++j) bw[(size_t)j * 16] = (_Float16)wp[j];
        } else {
            // guarded path: clamped-index unconditional loads + cndmask selects
#pragma unroll
            for (int j = 0; j < 4; ++j) {
                const int kk  = k0 + a_kq + j;
                const bool kin = (kk < K);
                const int ks  = kin ? kk : 0;
                float v0 = A[a_row0 + ks];
                float v1 = A[a_row1 + ks];
                aw0[j] = (_Float16)((a_ok0 && kin) ? v0 : 0.0f);
                aw1[j] = (_Float16)((a_ok1 && kin) ? v1 : 0.0f);
            }
            const int kk  = k0 + b_k;
            const bool kok = (kk < K);
            const size_t b_row = (size_t)(kok ? kk : 0) * N;
#pragma unroll
            for (int j = 0; j < 16; ++j) {
                const int n  = nblk + b_nc * 16 + j;
                const bool ok = kok && (n < N);
                const int ns = (n < N) ? n : 0;
                float v = W[b_row + ns];
                bw[(size_t)j * 16] = (_Float16)(ok ? v : 0.0f);
            }
        }
    };

    v8f acc0 = {};
    v8f acc1 = {};
    int cur = 0;

    stage(0, 0);
    __syncthreads();

    for (int k0 = 0; k0 < K; k0 += 32) {
        // fragment reads: contiguous 32B LDS reads (2x ds_load_b128 each)
        const v16h a0 = *reinterpret_cast<const v16h*>(&sA[cur][0][lane][0]);
        const v16h a1 = *reinterpret_cast<const v16h*>(&sA[cur][1][lane][0]);
        const v16h b  = *reinterpret_cast<const v16h*>(&sB[cur][wave][lane][0]);

        if (k0 + 32 < K) {
            if (k0 + 64 < K) {           // near-cache prefetch of following rows
                __builtin_prefetch(&A[a_row0 + k0 + 64], 0, 3);
                __builtin_prefetch(&A[a_row1 + k0 + 64], 0, 3);
            }
            stage(cur ^ 1, k0 + 32);     // overlap next-tile loads with WMMA
        }

        // two WMMAs share one staged B fragment
        acc0 = __builtin_amdgcn_wmma_f32_16x16x32_f16(false, a0, false, b,
                                                      (short)0, acc0, false, false);
        acc1 = __builtin_amdgcn_wmma_f32_16x16x32_f16(false, a1, false, b,
                                                      (short)0, acc1, false, false);
        __syncthreads();                 // one barrier per K-step
        cur ^= 1;
    }

    // ---- epilogue: C layout (ISA 7.12.2): vgpr r -> row r+8*half_sel, col l16
    const int n0   = nblk + wave * 16;
    const int ncol = n0 + l16;
    float bv = 0.0f;
    if (bias) bv = bias[(ncol < N) ? ncol : 0];
    if (fullM && fullN) {
#pragma unroll
        for (int r = 0; r < 8; ++r) {
            float v = acc0[r] + bv;
            if (act == 1) v = fmaxf(v, 0.0f);
            C[(size_t)(m0 + r + 8 * half_sel) * ldc + ncol] = v;
        }
#pragma unroll
        for (int r = 0; r < 8; ++r) {
            float v = acc1[r] + bv;
            if (act == 1) v = fmaxf(v, 0.0f);
            C[(size_t)(m0 + 16 + r + 8 * half_sel) * ldc + ncol] = v;
        }
    } else {
#pragma unroll
        for (int r = 0; r < 8; ++r) {
            const int mrow = m0 + r + 8 * half_sel;
            if (mrow < M && ncol < N) {
                float v = acc0[r] + bv;
                if (act == 1) v = fmaxf(v, 0.0f);
                C[(size_t)mrow * ldc + ncol] = v;
            }
        }
#pragma unroll
        for (int r = 0; r < 8; ++r) {
            const int mrow = m0 + 16 + r + 8 * half_sel;
            if (mrow < M && ncol < N) {
                float v = acc1[r] + bv;
                if (act == 1) v = fmaxf(v, 0.0f);
                C[(size_t)mrow * ldc + ncol] = v;
            }
        }
    }
}

// ---------------------------------------------------------------------------
// dt = softplus(proj[..., DINNER+CONVDIM : ] + dt_bias)
// ---------------------------------------------------------------------------
__global__ void dt_kernel(const float* __restrict__ proj,
                          const float* __restrict__ dtb,
                          float* __restrict__ dtout)
{
    const int idx = blockIdx.x * blockDim.x + threadIdx.x;
    if (idx >= BATCH * SEQLEN * NHEADS) return;
    const int h  = idx % NHEADS;
    const int bl = idx / NHEADS;
    const float v = proj[(size_t)bl * DINPROJ + DINNER + CONVDIM + h] + dtb[h];
    dtout[idx] = (v > 20.0f) ? v : log1pf(__expf(v));
}

// ---------------------------------------------------------------------------
// depthwise causal conv (width 4, left-pad 3) over xBC slice of proj, + SiLU
// ---------------------------------------------------------------------------
__global__ void conv_silu_kernel(const float* __restrict__ proj,
                                 const float* __restrict__ cw,  // CONVDIM x DCONV
                                 const float* __restrict__ cb,  // CONVDIM
                                 float* __restrict__ xbc)       // B*L x CONVDIM
{
    const int idx = blockIdx.x * blockDim.x + threadIdx.x;
    if (idx >= BATCH * SEQLEN * CONVDIM) return;
    const int c  = idx % CONVDIM;
    const int bl = idx / CONVDIM;
    const int l  = bl % SEQLEN;
    const int b  = bl / SEQLEN;
    float s = cb[c];
#pragma unroll
    for (int k = 0; k < DCONV; ++k) {
        const int ls = l - (DCONV - 1) + k;
        const bool ok = (ls >= 0);
        const int lss = ok ? ls : 0;
        float v = proj[((size_t)(b * SEQLEN + lss)) * DINPROJ + DINNER + c];
        s += (ok ? v : 0.0f) * cw[c * DCONV + k];
    }
    xbc[idx] = siluf_(s);
}

// ---------------------------------------------------------------------------
// Selective-state scan. One block per (batch, head); 256 threads; 8 waves.
// State h[64p][64n] lives in registers: thread t owns p = t>>2, n = (t&3)*16..+16.
// Dot product over n reduced with 2 shfl_xor hops (4 lanes per p, wave32).
// ---------------------------------------------------------------------------
__global__ void ssm_scan_kernel(const float* __restrict__ xbc,
                                const float* __restrict__ dtb,
                                const float* __restrict__ A_log,
                                const float* __restrict__ Dp,
                                float* __restrict__ y)          // B*L x DINNER
{
    const int b = blockIdx.x / NHEADS;
    const int h = blockIdx.x % NHEADS;
    const int t = threadIdx.x;          // 256
    const int p  = t >> 2;
    const int nb = (t & 3) * 16;

    float st[16];
#pragma unroll
    for (int j = 0; j < 16; ++j) st[j] = 0.0f;

    __shared__ float sx[HEADDIM];
    __shared__ float sB[DSTATE];
    __shared__ float sC[DSTATE];

    const float Ah = -__expf(A_log[h]);
    const float Dh = Dp[h];

    for (int l = 0; l < SEQLEN; ++l) {
        const size_t base = ((size_t)(b * SEQLEN + l)) * CONVDIM;
        if (t < 64)        sx[t]        = xbc[base + h * HEADDIM + t];
        else if (t < 128)  sB[t - 64]   = xbc[base + DINNER + (t - 64)];
        else if (t < 192)  sC[t - 128]  = xbc[base + DINNER + DSTATE + (t - 128)];
        __syncthreads();

        const float dt  = dtb[(size_t)(b * SEQLEN + l) * NHEADS + h];
        const float dA  = __expf(dt * Ah);
        const float xp  = sx[p];
        const float dtx = dt * xp;

        float part = 0.0f;
#pragma unroll
        for (int j = 0; j < 16; ++j) {
            st[j] = st[j] * dA + dtx * sB[nb + j];
            part += st[j] * sC[nb + j];
        }
        part += __shfl_xor(part, 1, 32);
        part += __shfl_xor(part, 2, 32);
        if ((t & 3) == 0)
            y[((size_t)(b * SEQLEN + l)) * DINNER + h * HEADDIM + p] = part + Dh * xp;
        __syncthreads();
    }
}

// ---------------------------------------------------------------------------
// y = rmsnorm( y * silu(z) ) * nrm_w    (block per token; 256 thr x 2 elems)
// ---------------------------------------------------------------------------
__global__ void gate_rms_kernel(float* __restrict__ y,
                                const float* __restrict__ proj,
                                const float* __restrict__ nw)
{
    const int bl = blockIdx.x;
    const int t  = threadIdx.x;         // 256
    const size_t yb = (size_t)bl * DINNER;
    const size_t pb = (size_t)bl * DINPROJ;

    const float v0 = y[yb + t]       * siluf_(proj[pb + t]);
    const float v1 = y[yb + t + 256] * siluf_(proj[pb + t + 256]);

    __shared__ float red[256];
    red[t] = v0 * v0 + v1 * v1;
    __syncthreads();
    for (int s = 128; s > 0; s >>= 1) {
        if (t < s) red[t] += red[t + s];
        __syncthreads();
    }
    const float scale = rsqrtf(red[0] / (float)DINNER + 1e-5f);
    y[yb + t]       = v0 * scale * nw[t];
    y[yb + t + 256] = v1 * scale * nw[t + 256];
}

// ---------------------------------------------------------------------------
// h = layernorm(tmp + h) * g + b       (block per token; 256 thr = DMODEL)
// ---------------------------------------------------------------------------
__global__ void add_layernorm_kernel(const float* __restrict__ tmp,
                                     float* __restrict__ h,
                                     const float* __restrict__ g,
                                     const float* __restrict__ bb)
{
    const int bl = blockIdx.x;
    const int t  = threadIdx.x;         // 256 == DMODEL
    const size_t base = (size_t)bl * DMODEL;

    const float v = tmp[base + t] + h[base + t];
    __shared__ float r1[256];
    __shared__ float r2[256];
    r1[t] = v; r2[t] = v * v;
    __syncthreads();
    for (int s = 128; s > 0; s >>= 1) {
        if (t < s) { r1[t] += r1[t + s]; r2[t] += r2[t + s]; }
        __syncthreads();
    }
    const float mu  = r1[0] / (float)DMODEL;
    const float var = r2[0] / (float)DMODEL - mu * mu;
    h[base + t] = (v - mu) * rsqrtf(var + 1e-5f) * g[t] + bb[t];
}

// ---------------------------------------------------------------------------
extern "C" void kernel_launch(void* const* d_in, const int* in_sizes, int n_in,
                              void* d_out, int out_size, void* d_ws, size_t ws_size,
                              hipStream_t stream)
{
    (void)in_sizes; (void)n_in; (void)out_size; (void)ws_size;

    const float* x       = (const float*)d_in[0];
    const float* enc_W   = (const float*)d_in[1];
    const float* enc_b   = (const float*)d_in[2];
    const float* in_W    = (const float*)d_in[3];
    const float* conv_w  = (const float*)d_in[4];
    const float* conv_b  = (const float*)d_in[5];
    const float* dt_bias = (const float*)d_in[6];
    const float* A_log   = (const float*)d_in[7];
    const float* Dp      = (const float*)d_in[8];
    const float* nrm_w   = (const float*)d_in[9];
    const float* out_W   = (const float*)d_in[10];
    const float* ln_g    = (const float*)d_in[11];
    const float* ln_b    = (const float*)d_in[12];
    const float* l1_W    = (const float*)d_in[13];
    const float* l1_b    = (const float*)d_in[14];
    const float* l2_W    = (const float*)d_in[15];
    const float* l2_b    = (const float*)d_in[16];
    const float* l3_W    = (const float*)d_in[17];
    const float* l3_b    = (const float*)d_in[18];
    float* out = (float*)d_out;

    // ---- workspace carve-up (~46 MB) ----
    char* ws = (char*)d_ws;
    size_t off = 0;
    auto carve = [&](size_t bytes) {
        void* p = ws + off;
        off += (bytes + 255) & ~(size_t)255;
        return p;
    };
    const size_t BL = (size_t)BATCH * SEQLEN;
    float* hbuf = (float*)carve(BL * DMODEL  * sizeof(float));   // residual stream
    float* tmp  = (float*)carve(BL * DMODEL  * sizeof(float));   // out_proj result
    float* proj = (float*)carve(BL * DINPROJ * sizeof(float));   // zxbcdt
    float* xbc  = (float*)carve(BL * CONVDIM * sizeof(float));   // conv+silu output
    float* dtb  = (float*)carve(BL * NHEADS  * sizeof(float));   // softplus dt
    float* ybuf = (float*)carve(BL * DINNER  * sizeof(float));   // scan output / gated
    float* a1   = (float*)carve((size_t)BATCH * L1DIM * sizeof(float));
    float* a2   = (float*)carve((size_t)BATCH * L2DIM * sizeof(float));

    auto gemm = [&](const float* A, int lda, const float* W, const float* bias,
                    float* C, int ldc, int M, int N, int K, int act) {
        dim3 grid((M + 31) / 32, (N + 63) / 64);
        wmma_gemm_kernel<<<grid, 128, 0, stream>>>(A, lda, W, bias, C, ldc, M, N, K, act);
    };

    // encoder: h = x @ enc_W + enc_b
    gemm(x, DINPUT, enc_W, enc_b, hbuf, DMODEL, (int)BL, DMODEL, DINPUT, 0);

    for (int i = 0; i < NLAYERS; ++i) {
        // in_proj
        gemm(hbuf, DMODEL, in_W + (size_t)i * DMODEL * DINPROJ, nullptr,
             proj, DINPROJ, (int)BL, DINPROJ, DMODEL, 0);
        // dt softplus
        dt_kernel<<<(int)((BL * NHEADS + 255) / 256), 256, 0, stream>>>(
            proj, dt_bias + i * NHEADS, dtb);
        // depthwise causal conv + silu
        conv_silu_kernel<<<(int)((BL * CONVDIM + 255) / 256), 256, 0, stream>>>(
            proj, conv_w + (size_t)i * CONVDIM * DCONV, conv_b + (size_t)i * CONVDIM, xbc);
        // sequential SSM scan (32 independent (b,h) streams)
        ssm_scan_kernel<<<BATCH * NHEADS, 256, 0, stream>>>(
            xbc, dtb, A_log + i * NHEADS, Dp + i * NHEADS, ybuf);
        // gate with silu(z) + RMS norm
        gate_rms_kernel<<<(int)BL, 256, 0, stream>>>(ybuf, proj, nrm_w + (size_t)i * DINNER);
        // out_proj
        gemm(ybuf, DINNER, out_W + (size_t)i * DINNER * DMODEL, nullptr,
             tmp, DMODEL, (int)BL, DMODEL, DINNER, 0);
        // residual add + layernorm (in place on hbuf)
        add_layernorm_kernel<<<(int)BL, 256, 0, stream>>>(
            tmp, hbuf, ln_g + (size_t)i * DMODEL, ln_b + (size_t)i * DMODEL);
    }

    // head on last token: rows b are hbuf[b, L-1, :], so lda = SEQLEN*DMODEL
    const float* hlast = hbuf + (size_t)(SEQLEN - 1) * DMODEL;
    gemm(hlast, SEQLEN * DMODEL, l1_W, l1_b, a1, L1DIM, BATCH, L1DIM, DMODEL, 1);
    gemm(a1, L1DIM, l2_W, l2_b, a2, L2DIM, BATCH, L2DIM, L1DIM, 1);
    gemm(a2, L2DIM, l3_W, l3_b, out, DOUTPUT, BATCH, DOUTPUT, L2DIM, 0);
}
